// sarSSM_73581379715731
// MI455X (gfx1250) — matrix operand order repaired and checked by reading, not compiled, not run on hardware
//
#include <hip/hip_runtime.h>
#include <hip/hip_bf16.h>
#include <math.h>

// ---------------- problem constants ----------------
#define B_SZ   512
#define L_SEQ  10000
#define H_DIM  2
#define NHALF  32
#define LAYERS 4
#define T_CH   64
#define NC     157               // ceil(L_SEQ / T_CH)
#define L_PAD  (NC * T_CH)       // 10048
#define LDT    72                // padded LDS row stride (f16): 144 B, 16B-aligned, bank-scattering

typedef __attribute__((ext_vector_type(16))) _Float16 v16h;
typedef __attribute__((ext_vector_type(8)))  _Float16 v8h;
typedef __attribute__((ext_vector_type(4)))  _Float16 v4h;
typedef __attribute__((ext_vector_type(2)))  _Float16 v2h;
typedef __attribute__((ext_vector_type(8)))  float    v8f;

__device__ __forceinline__ v8f wmma_f16(v16h a, v16h b, v8f c) {
  // D = A(16x32 f16) * B(32x16 f16) + C(16x16 f32)
  return __builtin_amdgcn_wmma_f32_16x16x32_f16(false, a, false, b, (short)0, c, false, false);
}

// A fragment (16x32, f16) from row-major [ld] matrix (16B-aligned rows).
// Lane layout (ISA 7.12.2): lane<16: M=lane, K = {0..7, 16..23}; lane>=16: K = {8..15, 24..31}
__device__ __forceinline__ v16h load_frag_a(const _Float16* A, int ld, int base_m, int base_k, int lane) {
  int m  = base_m + (lane & 15);
  int hi = (lane >> 4) & 1;
  const _Float16* p = A + m * ld + base_k + hi * 8;
  v8h lo  = *(const v8h*)(p);        // K offsets hi*8 + 0..7
  v8h hi8 = *(const v8h*)(p + 16);   // K offsets hi*8 + 16..23
  return __builtin_shufflevector(lo, hi8, 0,1,2,3,4,5,6,7,8,9,10,11,12,13,14,15);
}

// B fragment (32x16, f16) from TRANSPOSED LDS tile uT[n][k] with row stride LDT.
// Lane layout: n = base_n + lane%16 ; K = base_k + (lane<16 ? {0..7,16..23} : {8..15,24..31})
// -> two contiguous ds_load_b128 per lane.
__device__ __forceinline__ v16h load_frag_b_t(const _Float16* uT, int base_k, int base_n, int lane) {
  int n  = base_n + (lane & 15);
  int k0 = base_k + ((lane >> 4) & 1) * 8;
  const _Float16* p = uT + n * LDT + k0;
  v8h lo  = *(const v8h*)(p);
  v8h hi8 = *(const v8h*)(p + 16);
  return __builtin_shufflevector(lo, hi8, 0,1,2,3,4,5,6,7,8,9,10,11,12,13,14,15);
}

// Stage a 64(k) x 128(b) f16 global tile (row stride B_SZ) into transposed LDS uT[b][k].
// Each thread owns 4x4 (k x b) micro-tiles: 4x v4h global loads (coalesced, 256 B/wave-op),
// in-register 4x4 transpose, 4x packed ds_store_b64.
__device__ __forceinline__ void stage_tile_T(const _Float16* __restrict__ gp, _Float16* __restrict__ uT) {
  for (int idx = threadIdx.x; idx < 32 * 16; idx += 256) {
    int b0 = (idx & 31) * 4;
    int s0 = (idx >> 5) * 4;
    v4h r0 = *(const v4h*)(gp + (size_t)(s0 + 0) * B_SZ + b0);
    v4h r1 = *(const v4h*)(gp + (size_t)(s0 + 1) * B_SZ + b0);
    v4h r2 = *(const v4h*)(gp + (size_t)(s0 + 2) * B_SZ + b0);
    v4h r3 = *(const v4h*)(gp + (size_t)(s0 + 3) * B_SZ + b0);
    v4h c0 = {r0[0], r1[0], r2[0], r3[0]};
    v4h c1 = {r0[1], r1[1], r2[1], r3[1]};
    v4h c2 = {r0[2], r1[2], r2[2], r3[2]};
    v4h c3 = {r0[3], r1[3], r2[3], r3[3]};
    *(v4h*)(uT + (b0 + 0) * LDT + s0) = c0;
    *(v4h*)(uT + (b0 + 1) * LDT + s0) = c1;
    *(v4h*)(uT + (b0 + 2) * LDT + s0) = c2;
    *(v4h*)(uT + (b0 + 3) * LDT + s0) = c3;
  }
}

// ---------------- kernel 1: input projection (B,L,3) @ W1^T + b1 -> u[h][l][b] (f16) ----------------
__global__ __launch_bounds__(256) void k_inproj(const float* __restrict__ x,
                                                const float* __restrict__ W1,
                                                const float* __restrict__ b1,
                                                _Float16* __restrict__ u) {
  size_t tid = (size_t)blockIdx.x * 256 + threadIdx.x;
  if (tid >= (size_t)L_PAD * B_SZ) return;
  int    b = (int)(tid % B_SZ);
  size_t l = tid / B_SZ;
  float u0 = 0.f, u1 = 0.f;
  if (l < L_SEQ) {
    const float* xp = x + ((size_t)b * L_SEQ + l) * 3;
    float x0 = xp[0], x1 = xp[1], x2 = xp[2];
    u0 = W1[0] * x0 + W1[1] * x1 + W1[2] * x2 + b1[0];
    u1 = W1[3] * x0 + W1[4] * x1 + W1[5] * x2 + b1[1];
  }
  u[l * B_SZ + b]                        = (_Float16)u0;
  u[(size_t)L_PAD * B_SZ + l * B_SZ + b] = (_Float16)u1;
}

// ---------------- kernel 2: per-layer S4D parameter matrices (tiny) ----------------
// Ktoe[h][t][s] = k(t-s) (t>=s) ; Bmat[h][n|n+32][s] = Re|Im lam^(63-s)
// Cmat[h][t][n|n+32] = 2Re|-2Im (Cn lam^(t+1)) ; lamT[h][n] = lam^64
__global__ __launch_bounds__(128) void k_params(const float* __restrict__ log_dt,
                                                const float* __restrict__ log_A_real,
                                                const float* __restrict__ A_imag,
                                                const float* __restrict__ Cp,
                                                _Float16* __restrict__ Ktoe,
                                                _Float16* __restrict__ Bmat,
                                                _Float16* __restrict__ Cmat,
                                                float* __restrict__ lamT) {
  int h = threadIdx.x >> 6;
  int t = threadIdx.x & 63;
  __shared__ float kvals[H_DIM][T_CH];
  float dt = expf(log_dt[h]);
  float kacc = 0.f;
  for (int n = 0; n < NHALF; ++n) {
    float Ar = -expf(log_A_real[h * NHALF + n]);
    float Ai = A_imag[h * NHALF + n];
    float dr = Ar * dt, di = Ai * dt;
    float er = expf(dr);
    float lr = er * cosf(di), li = er * sinf(di);
    float Cr = Cp[(h * NHALF + n) * 2 + 0], Ci = Cp[(h * NHALF + n) * 2 + 1];
    // Cn = (Cr + iCi) * (lambda - 1) / (Ar + iAi)
    float nr = lr - 1.0f, ni = li;
    float tr = Cr * nr - Ci * ni, ti = Cr * ni + Ci * nr;
    float den = Ar * Ar + Ai * Ai;
    float cnr = (tr * Ar + ti * Ai) / den, cni = (ti * Ar - tr * Ai) / den;
    // k(t) = 2 Re(Cn lambda^t)
    float ep = expf((float)t * dr);
    float pr = ep * cosf((float)t * di), pi = ep * sinf((float)t * di);
    kacc += 2.0f * (cnr * pr - cni * pi);
    // Bmat column s=t: lambda^(63-t)
    float eb = expf((float)(63 - t) * dr);
    float br = eb * cosf((float)(63 - t) * di), bi = eb * sinf((float)(63 - t) * di);
    Bmat[(h * T_CH + n) * T_CH + t]          = (_Float16)br;
    Bmat[(h * T_CH + n + NHALF) * T_CH + t]  = (_Float16)bi;
    // Cmat row t: Cn lambda^(t+1)
    float ec = expf((float)(t + 1) * dr);
    float qr = ec * cosf((float)(t + 1) * di), qi = ec * sinf((float)(t + 1) * di);
    Cmat[(h * T_CH + t) * T_CH + n]          = (_Float16)( 2.0f * (cnr * qr - cni * qi));
    Cmat[(h * T_CH + t) * T_CH + n + NHALF]  = (_Float16)(-2.0f * (cnr * qi + cni * qr));
    if (t == 0) {
      float eT = expf(64.0f * dr);
      lamT[(h * NHALF + n) * 2 + 0] = eT * cosf(64.0f * di);
      lamT[(h * NHALF + n) * 2 + 1] = eT * sinf(64.0f * di);
    }
  }
  kvals[h][t] = kacc;
  __syncthreads();
  for (int s = 0; s < T_CH; ++s)
    Ktoe[(h * T_CH + t) * T_CH + s] = (_Float16)((s <= t) ? kvals[h][t - s] : 0.0f);
}

// ---------------- kernel 3: WMMA state reduction  bstate = Bmat * U ----------------
// grid: (colblock 0..3, chunk 0..NC-1, h 0..1), 256 threads = 8 waves
// wave w: row-tile (w&3), col-tiles (w>>2)*4 .. +3
__global__ __launch_bounds__(256) void k_bstate(const _Float16* __restrict__ u,
                                                const _Float16* __restrict__ Bmat,
                                                float* __restrict__ bstate) {
  int cb = blockIdx.x, c = blockIdx.y, h = blockIdx.z;
  __shared__ __align__(16) _Float16 uT[128][LDT];   // 18 KB
  const _Float16* up = u + ((size_t)h * L_PAD + (size_t)c * T_CH) * B_SZ + cb * 128;
  stage_tile_T(up, &uT[0][0]);
  __syncthreads();
  int wave = threadIdx.x >> 5, lane = threadIdx.x & 31;
  int rt = (wave & 3) * 16;
  int jh = (wave >> 2) * 4;
  v16h a0 = load_frag_a(Bmat + h * T_CH * T_CH, T_CH, rt, 0,  lane);
  v16h a1 = load_frag_a(Bmat + h * T_CH * T_CH, T_CH, rt, 32, lane);
  float* outp = bstate + (((size_t)h * NC + c) * T_CH) * B_SZ;
  for (int j = jh; j < jh + 4; ++j) {
    v16h b0 = load_frag_b_t(&uT[0][0], 0,  j * 16, lane);
    v16h b1 = load_frag_b_t(&uT[0][0], 32, j * 16, lane);
    v8f acc = {};
    acc = wmma_f16(a0, b0, acc);
    acc = wmma_f16(a1, b1, acc);
    int n = lane & 15, hi = (lane >> 4) & 1;
#pragma unroll
    for (int r = 0; r < 8; ++r) {
      int row = rt + r + hi * 8;
      outp[(size_t)row * B_SZ + cb * 128 + j * 16 + n] = acc[r];
    }
  }
}

// ---------------- kernel 4: chunk scan  s(c+1) = lamT*s(c) + b(c) ; sprev[c] = s(c) (f16) --------
__global__ __launch_bounds__(256) void k_scan(const float* __restrict__ bstate,
                                              const float* __restrict__ lamT,
                                              _Float16* __restrict__ sprev) {
  int tid = blockIdx.x * 256 + threadIdx.x;   // H*NHALF*B = 32768 threads
  int b = tid & (B_SZ - 1);
  int n = (tid >> 9) & (NHALF - 1);
  int h = tid >> 14;
  float lr = lamT[(h * NHALF + n) * 2 + 0];
  float li = lamT[(h * NHALF + n) * 2 + 1];
  float sr = 0.f, si = 0.f;
  const float* bp = bstate + (size_t)h * NC * T_CH * B_SZ;
  _Float16*    sp = sprev  + (size_t)h * NC * T_CH * B_SZ;
  for (int c = 0; c < NC; ++c) {
    size_t base = (size_t)c * T_CH * B_SZ;
    sp[base + (size_t)n * B_SZ + b]           = (_Float16)sr;
    sp[base + (size_t)(n + NHALF) * B_SZ + b] = (_Float16)si;
    float br = bp[base + (size_t)n * B_SZ + b];
    float bi = bp[base + (size_t)(n + NHALF) * B_SZ + b];
    float nr = lr * sr - li * si + br;
    float ni = lr * si + li * sr + bi;
    sr = nr; si = ni;
  }
}

// ---------------- kernel 5: WMMA fused conv + broadcast + epilogue ----------------
// grid: (colblock 0..3, chunk 0..NC-1), 256 threads = 8 waves
// y = Ktoe*U + Cmat*sprev + D*u ; g = gelu(y) ; channel mix + GLU -> u_next (f16)
__global__ __launch_bounds__(256) void k_cross(const _Float16* __restrict__ u,
                                               const _Float16* __restrict__ sprev,
                                               const _Float16* __restrict__ Ktoe,
                                               const _Float16* __restrict__ Cmat,
                                               const float* __restrict__ Dvec,
                                               const float* __restrict__ Wout,
                                               const float* __restrict__ bout,
                                               _Float16* __restrict__ u_next) {
  int cb = blockIdx.x, c = blockIdx.y;
  __shared__ __align__(16) _Float16 uTs[H_DIM][128][LDT];  // 36 KB
  __shared__ __align__(16) _Float16 sTs[H_DIM][128][LDT];  // 36 KB
  __shared__ float gS[H_DIM][T_CH][128];                   // 64 KB
  for (int hh = 0; hh < H_DIM; ++hh) {
    const _Float16* up = u     + ((size_t)hh * L_PAD + (size_t)c * T_CH) * B_SZ + cb * 128;
    const _Float16* sp = sprev + (((size_t)hh * NC + c) * T_CH) * B_SZ + cb * 128;
    stage_tile_T(up, &uTs[hh][0][0]);
    stage_tile_T(sp, &sTs[hh][0][0]);
  }
  __syncthreads();
  int wave = threadIdx.x >> 5, lane = threadIdx.x & 31;
  int h = wave >> 2;
  int rt = (wave & 3) * 16;
  v16h aK0 = load_frag_a(Ktoe + h * T_CH * T_CH, T_CH, rt, 0,  lane);
  v16h aK1 = load_frag_a(Ktoe + h * T_CH * T_CH, T_CH, rt, 32, lane);
  v16h aC0 = load_frag_a(Cmat + h * T_CH * T_CH, T_CH, rt, 0,  lane);
  v16h aC1 = load_frag_a(Cmat + h * T_CH * T_CH, T_CH, rt, 32, lane);
  float Dh = Dvec[h];
  const _Float16* uu = u + ((size_t)h * L_PAD + (size_t)c * T_CH) * B_SZ + cb * 128;
  for (int j = 0; j < 8; ++j) {
    v16h bu0 = load_frag_b_t(&uTs[h][0][0], 0,  j * 16, lane);
    v16h bu1 = load_frag_b_t(&uTs[h][0][0], 32, j * 16, lane);
    v16h bs0 = load_frag_b_t(&sTs[h][0][0], 0,  j * 16, lane);
    v16h bs1 = load_frag_b_t(&sTs[h][0][0], 32, j * 16, lane);
    v8f acc = {};
    acc = wmma_f16(aK0, bu0, acc);
    acc = wmma_f16(aK1, bu1, acc);
    acc = wmma_f16(aC0, bs0, acc);
    acc = wmma_f16(aC1, bs1, acc);
    int n = lane & 15, hi = (lane >> 4) & 1;
#pragma unroll
    for (int r = 0; r < 8; ++r) {
      int row = rt + r + hi * 8;
      int col = j * 16 + n;
      float y = acc[r] + Dh * (float)uu[(size_t)row * B_SZ + col];
      gS[h][row][col] = 0.5f * y * (1.0f + erff(y * 0.70710678118f));   // exact gelu
    }
  }
  __syncthreads();
  for (int idx = threadIdx.x; idx < T_CH * 64; idx += 256) {
    int t  = idx >> 6;
    int b2 = (idx & 63) * 2;
    v2h o_lo, o_hi;
#pragma unroll
    for (int i = 0; i < 2; ++i) {
      float g0 = gS[0][t][b2 + i], g1 = gS[1][t][b2 + i];
      float o0 = Wout[0] * g0 + Wout[1] * g1 + bout[0];
      float o1 = Wout[2] * g0 + Wout[3] * g1 + bout[1];
      float o2 = Wout[4] * g0 + Wout[5] * g1 + bout[2];
      float o3 = Wout[6] * g0 + Wout[7] * g1 + bout[3];
      float s2 = 1.0f / (1.0f + expf(-o2));
      float s3 = 1.0f / (1.0f + expf(-o3));
      o_lo[i] = (_Float16)(o0 * s2);
      o_hi[i] = (_Float16)(o1 * s3);
    }
    size_t off = ((size_t)c * T_CH + t) * B_SZ + cb * 128 + b2;
    *(v2h*)(u_next + off)                        = o_lo;
    *(v2h*)(u_next + (size_t)L_PAD * B_SZ + off) = o_hi;
  }
}

// ---------------- kernel 6: output projection u @ W2^T + b2 -> (B,L,2) f32 ----------------
__global__ __launch_bounds__(256) void k_outproj(const _Float16* __restrict__ u,
                                                 const float* __restrict__ W2,
                                                 const float* __restrict__ b2,
                                                 float* __restrict__ out) {
  size_t tid = (size_t)blockIdx.x * 256 + threadIdx.x;
  if (tid >= (size_t)L_SEQ * B_SZ) return;
  int    b = (int)(tid % B_SZ);
  size_t l = tid / B_SZ;
  float u0 = (float)u[l * B_SZ + b];
  float u1 = (float)u[(size_t)L_PAD * B_SZ + l * B_SZ + b];
  out[((size_t)b * L_SEQ + l) * 2 + 0] = W2[0] * u0 + W2[1] * u1 + b2[0];
  out[((size_t)b * L_SEQ + l) * 2 + 1] = W2[2] * u0 + W2[3] * u1 + b2[1];
}

// ---------------- host ----------------
extern "C" void kernel_launch(void* const* d_in, const int* in_sizes, int n_in,
                              void* d_out, int out_size, void* d_ws, size_t ws_size,
                              hipStream_t stream) {
  const float* x          = (const float*)d_in[0];
  const float* W1         = (const float*)d_in[1];
  const float* b1         = (const float*)d_in[2];
  const float* W2         = (const float*)d_in[3];
  const float* b2         = (const float*)d_in[4];
  const float* log_dt     = (const float*)d_in[5];
  const float* log_A_real = (const float*)d_in[6];
  const float* A_imag     = (const float*)d_in[7];
  const float* Cc         = (const float*)d_in[8];
  const float* Dv         = (const float*)d_in[9];
  const float* Wout       = (const float*)d_in[10];
  const float* bout       = (const float*)d_in[11];
  float* out = (float*)d_out;

  const size_t NBIG = (size_t)H_DIM * L_PAD * B_SZ;   // 10,289,152 elements
  float*    bstate = (float*)d_ws;                    // f32  (41.2 MB)
  _Float16* u_a    = (_Float16*)(bstate + NBIG);      // f16  (20.6 MB)
  _Float16* u_b    = u_a + NBIG;                      // f16
  _Float16* sprevB = u_b + NBIG;                      // f16
  _Float16* Ktoe   = sprevB + NBIG;
  _Float16* Bmat   = Ktoe + H_DIM * T_CH * T_CH;
  _Float16* Cmat   = Bmat + H_DIM * T_CH * T_CH;
  float*    lamT   = (float*)(Cmat + H_DIM * T_CH * T_CH);

  size_t nin = (size_t)L_PAD * B_SZ;
  k_inproj<<<dim3((unsigned)((nin + 255) / 256)), 256, 0, stream>>>(x, W1, b1, u_a);

  _Float16* ucur = u_a;
  _Float16* unxt = u_b;
  for (int layer = 0; layer < LAYERS; ++layer) {
    k_params<<<1, 128, 0, stream>>>(log_dt + layer * H_DIM,
                                    log_A_real + layer * H_DIM * NHALF,
                                    A_imag + layer * H_DIM * NHALF,
                                    Cc + layer * H_DIM * NHALF * 2,
                                    Ktoe, Bmat, Cmat, lamT);
    dim3 gi(4, NC, H_DIM);
    k_bstate<<<gi, 256, 0, stream>>>(ucur, Bmat, bstate);
    k_scan<<<128, 256, 0, stream>>>(bstate, lamT, sprevB);
    dim3 gc(4, NC);
    k_cross<<<gc, 256, 0, stream>>>(ucur, sprevB, Ktoe, Cmat,
                                    Dv + layer * H_DIM, Wout + layer * 2 * H_DIM * H_DIM,
                                    bout + layer * 2 * H_DIM, unxt);
    _Float16* t = ucur; ucur = unxt; unxt = t;
  }

  size_t nout = (size_t)L_SEQ * B_SZ;
  k_outproj<<<dim3((unsigned)((nout + 255) / 256)), 256, 0, stream>>>(ucur, W2, b2, out);
}
